// Attention_39402029973930
// MI455X (gfx1250) — matrix operand order repaired and checked by reading, not compiled, not run on hardware
//
#include <hip/hip_runtime.h>
#include <hip/hip_bf16.h>
#include <hip/hip_fp16.h>

typedef _Float16 v16h __attribute__((ext_vector_type(16)));
typedef float    v8f  __attribute__((ext_vector_type(8)));

#define H_    32
#define D_    128
#define DIM_  4096
#define T_    16384
#define TB_   2048
#define BS_   8
#define MB_   145
#define NSEL_ 161          // 8 sink + 8 window + 145 top
#define NTOK_ 1288         // NSEL_ * BS_
#define NTILE_ 81          // ceil(1288/16)
#define NPAD_ (NTILE_*16)  // 1296
#define SINKB_ 8
#define WINS_  2040
#define SCALE_ 0.08838834764831845f   // 1/sqrt(128)

__device__ __forceinline__ float waveReduceSum(float v) {
#pragma unroll
  for (int m = 16; m > 0; m >>= 1) v += __shfl_xor(v, m, 32);
  return v;
}

// ---------------- streaming GEMV: y[row] = dot(W[row,:], x) -----------------
// one wave per row, float4 (global_load_b128) coalesced; pure bandwidth.
__global__ void __launch_bounds__(256) gemv_kernel(
    const float* __restrict__ W, const float* __restrict__ x,
    float* __restrict__ y, int rows, int cols) {
  int wave = threadIdx.x >> 5, lane = threadIdx.x & 31;
  int row = blockIdx.x * 8 + wave;
  if (row >= rows) return;
  const float4* wr = (const float4*)(W + (size_t)row * cols);
  const float4* xr = (const float4*)x;
  int n4 = cols >> 2;
  float acc = 0.f;
#pragma unroll 4
  for (int i = lane; i < n4; i += 32) {
    float4 a = wr[i], b = xr[i];
    acc += a.x*b.x + a.y*b.y + a.z*b.z + a.w*b.w;
  }
  acc = waveReduceSum(acc);
  if (lane == 0) y[row] = acc;
}

// ---------------- RoPE on q and k (new token) -------------------------------
// ws layout (floats): [0,12288) qkv ; [12288,16384) q_rope ; [16384,20480) k_rope
__global__ void rope_kernel(float* ws, const float* __restrict__ freqs) {
  int idx = blockIdx.x * blockDim.x + threadIdx.x;   // 0..4095
  if (idx >= 2 * H_ * (D_ / 2)) return;
  int which = idx >> 11;            // 0 = q, 1 = k
  int r = idx & 2047;
  int h = r >> 6, j = r & 63;
  const float* src = ws + which * DIM_ + h * D_;
  float xr = src[2*j], xi = src[2*j+1];
  float fr = freqs[2*j], fi = freqs[2*j+1];
  float* dst = ws + 3*DIM_ + which * DIM_ + h * D_;
  dst[2*j]   = xr*fr - xi*fi;
  dst[2*j+1] = xi*fr + xr*fi;
}

// ---------------- block-mean scores: one wave per (head, block) -------------
// lane owns cols [lane*4, lane*4+4); q lives in registers; each iteration is
// one contiguous 512B wave-wide row read -> optimal for the 256MB cache scan.
__global__ void __launch_bounds__(256) scores_kernel(
    const float* __restrict__ kc, const float* __restrict__ ws,
    float* __restrict__ scores, const int* __restrict__ pos) {
  int wave = threadIdx.x >> 5, lane = threadIdx.x & 31;
  int id = blockIdx.x * 8 + wave;        // 0..65535
  int h = id >> 11, b = id & (TB_ - 1);
  int tnew = *pos;
  int col = lane * 4;
  const float* krope = ws + 4*DIM_ + h*D_;
  float4 qv = *(const float4*)(ws + 3*DIM_ + h*D_ + col);
  const float* kbase = kc + ((size_t)h * T_ + (size_t)b * BS_) * D_;
  float acc = 0.f;
#pragma unroll
  for (int it = 0; it < BS_; ++it) {
    int tok = b * BS_ + it;
    const float* krow = (tok == tnew) ? krope : (kbase + (size_t)it * D_);
    float4 kv = *(const float4*)(krow + col);
    acc += kv.x*qv.x + kv.y*qv.y + kv.z*qv.z + kv.w*qv.w;
  }
  acc = waveReduceSum(acc);
  if (lane == 0) scores[h * TB_ + b] = acc * (SCALE_ / 8.0f);
}

// ---------------- per-head top-145 with sink/window exclusion ---------------
__global__ void __launch_bounds__(256) topk_kernel(
    const float* __restrict__ scores, int* __restrict__ bi) {
  __shared__ float sc[TB_];
  __shared__ float wv[8];
  __shared__ int   wi[8];
  int h = blockIdx.x, tid = threadIdx.x;
  int lane = tid & 31, wave = tid >> 5;
  for (int i = tid; i < TB_; i += 256) {
    bool excl = (i < SINKB_) || (i >= WINS_);
    sc[i] = excl ? -1e30f : scores[h * TB_ + i];
  }
  if (tid < SINKB_)      bi[h*NSEL_ + tid] = tid;
  else if (tid < 16)     bi[h*NSEL_ + tid] = WINS_ + (tid - 8);
  __syncthreads();
  for (int it = 0; it < MB_; ++it) {
    float bv = -3e30f; int bidx = TB_;
#pragma unroll
    for (int s = 0; s < TB_/256; ++s) {
      int i = tid + s*256;
      float v = sc[i];
      if (v > bv) { bv = v; bidx = i; }      // strided asc -> smallest idx kept
    }
#pragma unroll
    for (int m = 16; m > 0; m >>= 1) {       // wave reduce: desc value, asc idx
      float ov = __shfl_xor(bv, m, 32);
      int   oi = __shfl_xor(bidx, m, 32);
      if (ov > bv || (ov == bv && oi < bidx)) { bv = ov; bidx = oi; }
    }
    if (lane == 0) { wv[wave] = bv; wi[wave] = bidx; }
    __syncthreads();
    if (tid == 0) {
      float best = wv[0]; int besti = wi[0];
#pragma unroll
      for (int w = 1; w < 8; ++w)
        if (wv[w] > best || (wv[w] == best && wi[w] < besti)) { best = wv[w]; besti = wi[w]; }
      bi[h*NSEL_ + 16 + it] = besti;
      sc[besti] = -3e30f;
    }
    __syncthreads();
  }
}

// ---------------- per-head attention: WMMA q·K tiles, softmax, p·V ----------
__global__ void __launch_bounds__(128) attn_kernel(
    const float* __restrict__ kc, const float* __restrict__ vc,
    const float* __restrict__ ws, const int* __restrict__ bi,
    const int* __restrict__ pos, float* __restrict__ outh) {
  __shared__ _Float16 ktile[4][16 * D_];   // 4 waves x (16 tokens x 128) f16
  __shared__ _Float16 q16[D_];
  __shared__ float att[NPAD_];
  __shared__ float red[128];
  __shared__ int   bsel[NSEL_];
  int h = blockIdx.x, tid = threadIdx.x;
  int wave = tid >> 5, lane = tid & 31;
  int tnew = *pos;
  const float* qrow  = ws + 3*DIM_ + h*D_;
  const float* krope = ws + 4*DIM_ + h*D_;
  const float* vnew  = ws + 2*DIM_ + h*D_;
  if (tid < D_) q16[tid] = (_Float16)qrow[tid];
  for (int i = tid; i < NSEL_; i += 128) bsel[i] = bi[h*NSEL_ + i];
  __syncthreads();

  // ---- scores via v_wmma_f32_16x16x32_f16, 16 tokens per tile per wave ----
  // 81 tiles exactly cover att[0..1295]; every tile writes 16 scores
  // unconditionally (no divergent guards); pads re-masked after the loop.
  for (int t = wave; t < NTILE_; t += 4) {
    _Float16* kt = ktile[wave];
    {   // coalesced stage of 16 gathered K rows -> LDS f16
      int r = lane >> 1;
      int n = t*16 + r;
      int nn = (n < NTOK_) ? n : 0;        // pad slots load token 0 (masked later)
      int tok = bsel[nn >> 3] * BS_ + (nn & 7);
      const float* src = (tok == tnew) ? krope : (kc + ((size_t)h*T_ + tok)*D_);
      int c0 = (lane & 1) * 64;
#pragma unroll
      for (int i = 0; i < 16; ++i) {
        float4 v = *(const float4*)(src + c0 + i*4);
        int o = r*D_ + c0 + i*4;
        kt[o+0] = (_Float16)v.x; kt[o+1] = (_Float16)v.y;
        kt[o+2] = (_Float16)v.z; kt[o+3] = (_Float16)v.w;
      }
    }
    int m = lane & 15, hh = lane >> 4;
    v8f c = {};
#pragma unroll
    for (int kb = 0; kb < D_; kb += 32) {
      v16h a, bq;
#pragma unroll
      for (int j = 0; j < 8; ++j) {        // ISA 16-bit A 16x32 lane layout
        int k0 = (j < 4 ? 2*j : 16 + 2*(j-4)) + 8*hh;
        a[2*j]   = kt[m*D_ + kb + k0];
        a[2*j+1] = kt[m*D_ + kb + k0 + 1];
      }
#pragma unroll
      for (int e = 0; e < 16; ++e)         // B: q broadcast across N columns
        bq[e] = q16[kb + (e < 8 ? e : e + 8) + 8*hh];
      c = __builtin_amdgcn_wmma_f32_16x16x32_f16(
              false, a, false, bq, (short)0, c, false, false);
    }
    if ((lane & 15) == 0) {                // lanes 0/16: C VGPR r -> M = r + 8*hh
      int nb = t*16 + hh*8;
#pragma unroll
      for (int r = 0; r < 8; ++r) att[nb + r] = c[r] * SCALE_;
    }
  }
  __syncthreads();
  if (tid < NPAD_ - NTOK_) att[NTOK_ + tid] = -1e30f;   // mask the 8 pad slots
  __syncthreads();

  // ---- softmax over 1296 ----
  float lm = -1e30f;
  for (int i = tid; i < NPAD_; i += 128) lm = fmaxf(lm, att[i]);
  red[tid] = lm; __syncthreads();
  for (int s = 64; s > 0; s >>= 1) { if (tid < s) red[tid] = fmaxf(red[tid], red[tid+s]); __syncthreads(); }
  float gm = red[0]; __syncthreads();
  float ls = 0.f;
  for (int i = tid; i < NPAD_; i += 128) { float e = __expf(att[i] - gm); att[i] = e; ls += e; }
  red[tid] = ls; __syncthreads();
  for (int s = 64; s > 0; s >>= 1) { if (tid < s) red[tid] += red[tid+s]; __syncthreads(); }
  float inv = 1.0f / red[0];
  __syncthreads();

  // ---- out[d] = sum_n p[n] * V[tok(n), d] : fully coalesced 512B rows ----
  int d = tid;                             // blockDim.x == 128 == D_
  float acc = 0.f;
#pragma unroll 4
  for (int n = 0; n < NTOK_; ++n) {
    int tok = bsel[n >> 3] * BS_ + (n & 7);
    float v = (tok == tnew) ? vnew[d] : vc[((size_t)h*T_ + tok)*D_ + d];
    acc += att[n] * v;
  }
  outh[h*D_ + d] = acc * inv;
}

extern "C" void kernel_launch(void* const* d_in, const int* in_sizes, int n_in,
                              void* d_out, int out_size, void* d_ws, size_t ws_size,
                              hipStream_t stream) {
  (void)in_sizes; (void)n_in; (void)out_size; (void)ws_size;
  const float* x     = (const float*)d_in[0];
  const float* freqs = (const float*)d_in[1];
  const float* wqkv  = (const float*)d_in[2];
  const float* wo    = (const float*)d_in[3];
  const float* kc    = (const float*)d_in[4];
  const float* vc    = (const float*)d_in[5];
  const int*   pos   = (const int*)d_in[6];

  float* ws     = (float*)d_ws;
  float* qkv    = ws;              // 12288 f32 (q|k|v) ; rope out at 12288/16384
  float* scores = ws + 20480;      // 32*2048 f32
  int*   bi     = (int*)(ws + 86016); // 32*161 int
  float* outh   = ws + 91200;      // 32*128 f32     (total ~381 KB)

  gemv_kernel  <<<1536, 256, 0, stream>>>(wqkv, x, qkv, 3*DIM_, DIM_); // 201 MB
  rope_kernel  <<<16,   256, 0, stream>>>(ws, freqs);
  scores_kernel<<<8192, 256, 0, stream>>>(kc, ws, scores, pos);        // 256 MB
  topk_kernel  <<<32,   256, 0, stream>>>(scores, bi);
  attn_kernel  <<<32,   128, 0, stream>>>(kc, vc, ws, bi, pos, outh);  // 42 MB + WMMA
  gemv_kernel  <<<512,  256, 0, stream>>>(wo, outh, (float*)d_out, DIM_, DIM_); // 67 MB
}
